// NetworkODEModel_24429773979863
// MI455X (gfx1250) — matrix operand order repaired and checked by reading, not compiled, not run on hardware
//
#include <hip/hip_runtime.h>
#include <cstdint>

typedef __attribute__((ext_vector_type(16))) _Float16 v16h;
typedef __attribute__((ext_vector_type(8)))  float    v8f;

union V16H { v16h v; _Float16 e[16]; };
union V8F  { v8f  v; float    e[8];  };

// NEG_SLOPE = 0.01 < 1  =>  leaky(x) = max(x, 0.01*x)  (mul+max, no cmp/cndmask)
__device__ __forceinline__ float leaky(float v) { return fmaxf(v, 0.01f * v); }

// ---------------------------------------------------------------------------
// Phase 1: node MLP (D->HN->HN->D) into out, plus U = x@cW1[:16], V = x@cW1[16:]
// One block per (b,n) node, 64 threads. ~34 MFLOP total, negligible.
// ---------------------------------------------------------------------------
__global__ __launch_bounds__(64) void node_uv_kernel(
    const float* __restrict__ x,
    const float* __restrict__ nW1, const float* __restrict__ nb1,
    const float* __restrict__ nW2, const float* __restrict__ nb2,
    const float* __restrict__ nWo, const float* __restrict__ nbo,
    const float* __restrict__ cW1,
    float* __restrict__ U, float* __restrict__ V,
    float* __restrict__ out)
{
    __shared__ float xr[16];
    __shared__ float h1[64];
    __shared__ float h2[64];
    const int node = blockIdx.x;
    const int t = threadIdx.x;
    if (t < 16) xr[t] = x[node * 16 + t];
    __syncthreads();
    float a = nb1[t], u = 0.f, vv = 0.f;
#pragma unroll
    for (int d = 0; d < 16; ++d) {
        const float xv = xr[d];
        a  += xv * nW1[d * 64 + t];
        u  += xv * cW1[d * 64 + t];
        vv += xv * cW1[(16 + d) * 64 + t];
    }
    h1[t] = leaky(a);
    U[node * 64 + t] = u;
    V[node * 64 + t] = vv;
    __syncthreads();
    float a2 = nb2[t];
#pragma unroll 8
    for (int k = 0; k < 64; ++k) a2 += h1[k] * nW2[k * 64 + t];
    h2[t] = leaky(a2);
    __syncthreads();
    if (t < 16) {
        float o = nbo[t];
#pragma unroll 8
        for (int k = 0; k < 64; ++k) o += h2[k] * nWo[k * 16 + t];
        out[node * 16 + t] = o;
    }
}

// ---------------------------------------------------------------------------
// Phase 2: coupling network, fused, with async double-buffered V staging.
// Block = (b, 8 i's), wave w handles i = i0+w. Per j-tile of 16:
//   C1^T(k,j) = leaky(U_i[k] + V_j[k] + cb1[k])   -> WMMA B operand (regs)
//   C2^T      = cW2^T @ C1^T + cb2                -> 8x v_wmma_f32_16x16x32_f16
//                                                    (bias preloaded as C operand)
//   sacc     += A(i,j) * leaky(C2^T)              -> per-lane FMA (C/D lane N = j)
// V tile jt+1 streamed with global_load_async_to_lds_b128 while tile jt computes.
// Epilogue: s = xor-shuffle reduce, out += s @ cWo + (sum_j A(i,j)) * cbo.
// ---------------------------------------------------------------------------
__global__ __launch_bounds__(256) void coupling_kernel(
    const float* __restrict__ A_p,
    const float* __restrict__ cb1,
    const float* __restrict__ cW2, const float* __restrict__ cb2,
    const float* __restrict__ cWo, const float* __restrict__ cbo,
    const float* __restrict__ U,  const float* __restrict__ Vg,
    float* __restrict__ out)
{
    __shared__ float sA[8][256];                        // adjacency rows for the 8 i's
    __shared__ float sUC[8][64];                        // U(b,i,:) + cb1
    __shared__ __align__(16) float sV[2][16][68];       // V j-tile, double buffered, padded
    __shared__ float sS[8][64];                         // reduced s vectors

    const int tid  = threadIdx.x;
    const int lane = tid & 31;
    const int wv   = tid >> 5;
    const int half = lane >> 4;
    const int l16  = lane & 15;
    const int b    = blockIdx.x >> 5;
    const int i0   = (blockIdx.x & 31) * 8;
    const int i    = i0 + wv;

    // adjacency: sigmoid(A_p - eye/eps) * (1-eye)  ->  off-diag sigmoid, diag 0
#pragma unroll
    for (int k = 0; k < 8; ++k) {
        const int idx = tid + k * 256;
        const int ii = idx >> 8, jj = idx & 255;
        const float ap = A_p[(i0 + ii) * 256 + jj];
        sA[ii][jj] = ((i0 + ii) == jj) ? 0.f : 1.f / (1.f + __expf(-ap));
    }
#pragma unroll
    for (int k = 0; k < 2; ++k) {
        const int idx = tid + k * 256;
        const int ii = idx >> 6, f = idx & 63;
        sUC[ii][f] = U[(size_t)(b * 256 + i0 + ii) * 64 + f] + cb1[f];
    }

    // A operands = cW2^T in registers, 16-bit A layout:
    // K(v,s,half) = (v<4 ? 2v+s : 16+2(v-4)+s) + 8*half ; M = mt*16 + l16
    V16H wA[4][2];
#pragma unroll
    for (int mt = 0; mt < 4; ++mt)
#pragma unroll
        for (int kk = 0; kk < 2; ++kk)
#pragma unroll
            for (int v = 0; v < 8; ++v)
#pragma unroll
                for (int s = 0; s < 2; ++s) {
                    const int K  = (v < 4 ? 2 * v : 16 + 2 * (v - 4)) + 8 * half + s;
                    const int kg = K + 32 * kk;
                    const int m  = mt * 16 + l16;
                    wA[mt][kk].e[2 * v + s] = (_Float16)cW2[kg * 64 + m];
                }

    // bias registers (C-operand init): C/D layout M = mt*16 + 8*half + v
    float br[4][8];
#pragma unroll
    for (int mt = 0; mt < 4; ++mt)
#pragma unroll
        for (int v = 0; v < 8; ++v) br[mt][v] = cb2[mt * 16 + 8 * half + v];

    float sacc[4][8];
#pragma unroll
    for (int mt = 0; mt < 4; ++mt)
#pragma unroll
        for (int v = 0; v < 8; ++v) sacc[mt][v] = 0.f;
    float wsum = 0.f;

    // per-thread async copy slice: 16B chunk (j, f..f+3) of the 4KB tile
    const int e  = tid * 4;
    const int jc = e >> 6, fc = e & 63;
    const float* gsrc = Vg + (size_t)b * 256 * 64 + e;          // + jt*1024
    const unsigned ldsd0 = (unsigned)(uintptr_t)&sV[0][jc][fc]; // flat low 32 = LDS offset
    const unsigned ldsd1 = (unsigned)(uintptr_t)&sV[1][jc][fc];

    // prime the pipeline: tile 0 -> buffer 0
    asm volatile("global_load_async_to_lds_b128 %0, %1, off"
                 :: "v"(ldsd0), "v"(gsrc) : "memory");

    for (int jt = 0; jt < 16; ++jt) {
        const int p = jt & 1;
        asm volatile("s_wait_asynccnt 0x0" ::: "memory");  // my chunk of tile jt landed
        __syncthreads();                                   // whole tile jt visible; buf 1-p free
        if (jt < 15) {                                     // stream tile jt+1 under compute
            asm volatile("global_load_async_to_lds_b128 %0, %1, off"
                         :: "v"(p ? ldsd0 : ldsd1), "v"(gsrc + (jt + 1) * 1024) : "memory");
        }

        const float w = sA[wv][jt * 16 + l16];             // lane's j = jt*16 + l16
        wsum += w;

        // B operands = C1^T, 16-bit B layout: N = l16 (=j), K = (v<4 ? 2v : 16+2(v-4)) + s + 8*half + 32*kk
        V16H bop[2];
#pragma unroll
        for (int kk = 0; kk < 2; ++kk)
#pragma unroll
            for (int v = 0; v < 8; ++v) {
                const int f = (v < 4 ? 2 * v : 16 + 2 * (v - 4)) + 8 * half + 32 * kk;
                const float2 vf = *(const float2*)&sV[p][l16][f];
                const float2 uf = *(const float2*)&sUC[wv][f];
                bop[kk].e[2 * v + 0] = (_Float16)leaky(uf.x + vf.x);
                bop[kk].e[2 * v + 1] = (_Float16)leaky(uf.y + vf.y);
            }

#pragma unroll
        for (int mt = 0; mt < 4; ++mt) {
            V8F acc;
#pragma unroll
            for (int v = 0; v < 8; ++v) acc.e[v] = br[mt][v];   // bias via C operand
            acc.v = __builtin_amdgcn_wmma_f32_16x16x32_f16(
                false, wA[mt][0].v, false, bop[0].v, (short)0, acc.v, false, false);
            acc.v = __builtin_amdgcn_wmma_f32_16x16x32_f16(
                false, wA[mt][1].v, false, bop[1].v, (short)0, acc.v, false, false);
            // C/D layout: N = l16 (=j), M = mt*16 + v + 8*half (= output feature h)
#pragma unroll
            for (int v = 0; v < 8; ++v)
                sacc[mt][v] += w * leaky(acc.e[v]);
        }
    }

    // reduce sacc over the 16 lanes of each half (each (feature, j) lives in one lane)
#pragma unroll
    for (int mt = 0; mt < 4; ++mt)
#pragma unroll
        for (int v = 0; v < 8; ++v) {
            float t2 = sacc[mt][v];
            t2 += __shfl_xor(t2, 1);
            t2 += __shfl_xor(t2, 2);
            t2 += __shfl_xor(t2, 4);
            t2 += __shfl_xor(t2, 8);
            sacc[mt][v] = t2;
        }
    // row-sum of A(i,:): each j appears once per half -> full-wave reduce, halve
    wsum += __shfl_xor(wsum, 1);
    wsum += __shfl_xor(wsum, 2);
    wsum += __shfl_xor(wsum, 4);
    wsum += __shfl_xor(wsum, 8);
    wsum += __shfl_xor(wsum, 16);
    wsum *= 0.5f;

    if (l16 == 0) {
#pragma unroll
        for (int mt = 0; mt < 4; ++mt)
#pragma unroll
            for (int v = 0; v < 8; ++v)
                sS[wv][mt * 16 + 8 * half + v] = sacc[mt][v];
    }
    __syncthreads();

    if (lane < 16) {
        float o = cbo[lane] * wsum;
#pragma unroll 8
        for (int h = 0; h < 64; ++h) o += sS[wv][h] * cWo[h * 16 + lane];
        out[(size_t)(b * 256 + i) * 16 + lane] += o;   // node_out already stored
    }
}

extern "C" void kernel_launch(void* const* d_in, const int* in_sizes, int n_in,
                              void* d_out, int out_size, void* d_ws, size_t ws_size,
                              hipStream_t stream) {
    (void)in_sizes; (void)n_in; (void)out_size; (void)ws_size;
    const float* x   = (const float*)d_in[0];
    const float* A_p = (const float*)d_in[1];
    const float* nW1 = (const float*)d_in[2];
    const float* nb1 = (const float*)d_in[3];
    const float* nW2 = (const float*)d_in[4];
    const float* nb2 = (const float*)d_in[5];
    const float* nWo = (const float*)d_in[6];
    const float* nbo = (const float*)d_in[7];
    const float* cW1 = (const float*)d_in[8];
    const float* cb1 = (const float*)d_in[9];
    const float* cW2 = (const float*)d_in[10];
    const float* cb2 = (const float*)d_in[11];
    const float* cWo = (const float*)d_in[12];
    const float* cbo = (const float*)d_in[13];
    float* out = (float*)d_out;

    float* U = (float*)d_ws;            // 2048*64 f32
    float* V = U + 2048 * 64;           // 2048*64 f32 (1 MB total)

    node_uv_kernel<<<2048, 64, 0, stream>>>(x, nW1, nb1, nW2, nb2, nWo, nbo, cW1, U, V, out);
    coupling_kernel<<<256, 256, 0, stream>>>(A_p, cb1, cW2, cb2, cWo, cbo, U, V, out);
}
